// CoordAttMeanMax_53051436040385
// MI455X (gfx1250) — matrix operand order repaired and checked by reading, not compiled, not run on hardware
//
#include <hip/hip_runtime.h>
#include <math.h>

// Problem constants
#define NB   32
#define CC   256
#define HH   56
#define WWID 56
#define LL   112           // H + W (concat pooled length)
#define HW   3136          // 56*56
#define MIPC 8
#define EPSB 1e-5f

typedef __attribute__((ext_vector_type(16))) _Float16 v16h;
typedef __attribute__((ext_vector_type(8)))  _Float16 v8h;
typedef __attribute__((ext_vector_type(8)))  float    v8f;

// ---------------------------------------------------------------------------
// Kernel 1: pooling. One block per (n,c): stage 56x56 tile in padded LDS,
// produce row mean/max (l=0..55), col mean/max (l=56..111), and gap.
// This is the kernel that pulls x from HBM (~103 MB).
// ---------------------------------------------------------------------------
__global__ __launch_bounds__(256) void pool_kernel(
    const float* __restrict__ x,
    float* __restrict__ pmean, float* __restrict__ pmax,
    float* __restrict__ gap) {
  __shared__ float tile[HH * 57];   // 57-stride: conflict-free rows & cols
  __shared__ float rowsum[HH];
  const int nc  = blockIdx.x;
  const int tid = threadIdx.x;
  const float4* xb4 = (const float4*)(x + (size_t)nc * HW);
  for (int j = tid; j < HW / 4; j += 256) {
    __builtin_prefetch(xb4 + j + 256, 0, 0);
    float4 v = xb4[j];
    int h = (4 * j) / WWID, w0 = (4 * j) % WWID;   // never straddles a row
    float* t = &tile[h * 57 + w0];
    t[0] = v.x; t[1] = v.y; t[2] = v.z; t[3] = v.w;
  }
  __syncthreads();
  if (tid < HH) {
    float s = 0.f, m = -3.4e38f;
    #pragma unroll 8
    for (int w = 0; w < WWID; ++w) {
      float v = tile[tid * 57 + w];
      s += v; m = fmaxf(m, v);
    }
    pmean[(size_t)nc * LL + tid] = s * (1.f / WWID);
    pmax [(size_t)nc * LL + tid] = m;
    rowsum[tid] = s;
    float cs = 0.f, cm = -3.4e38f;
    #pragma unroll 8
    for (int h = 0; h < HH; ++h) {
      float v = tile[h * 57 + tid];
      cs += v; cm = fmaxf(cm, v);
    }
    pmean[(size_t)nc * LL + HH + tid] = cs * (1.f / HH);
    pmax [(size_t)nc * LL + HH + tid] = cm;
  }
  __syncthreads();
  if (tid == 0) {
    float tot = 0.f;
    for (int h = 0; h < HH; ++h) tot += rowsum[h];
    gap[nc] = tot * (1.f / HW);
  }
}

// ---------------------------------------------------------------------------
// Kernel 2: gate g = softmax(relu(gap@gw1.T)@gw2.T + gb2) -> (N,2)
// ---------------------------------------------------------------------------
__global__ __launch_bounds__(64) void gate_kernel(
    const float* __restrict__ gap,
    const float* __restrict__ gw1, const float* __restrict__ gw2,
    const float* __restrict__ gb2, float* __restrict__ g) {
  __shared__ float hid[64];
  __shared__ float logits[2];
  const int n = blockIdx.x, o = threadIdx.x;
  float a = 0.f;
  for (int c = 0; c < CC; ++c) a += gap[n * CC + c] * gw1[o * CC + c];
  hid[o] = fmaxf(a, 0.f);
  __syncthreads();
  if (o < 2) {
    float l = gb2[o];
    for (int k = 0; k < 64; ++k) l += hid[k] * gw2[o * 64 + k];
    logits[o] = l;
  }
  __syncthreads();
  if (o == 0) {
    float mx = fmaxf(logits[0], logits[1]);
    float e0 = __expf(logits[0] - mx), e1 = __expf(logits[1] - mx);
    float inv = 1.f / (e0 + e1);
    g[n * 2 + 0] = e0 * inv;
    g[n * 2 + 1] = e1 * inv;
  }
}

// ---------------------------------------------------------------------------
// WMMA fragment loaders — lane layouts as used here (wave32); A/B k order is contiguous, not the ISA table order.
// A (16x32 f16): lane%16 = M; koff = 16*(lane/16); element e <-> K = e + koff
//   (all e: K runs koff..koff+15)  ==> two contiguous 16-byte chunks.
// B (32x16 f16): lane%16 = N; lanes 0-15 hold K=0..15, lanes 16-31 K=16..31.
// C/D (16x16 f32): lane%16 = N; VGPR r -> M = 8*(lane/16) + r.
// ---------------------------------------------------------------------------
__device__ __forceinline__ v16h load_a_lds(const _Float16* __restrict__ base,
                                           int ldm, int lane) {
  // base -> element [0][k0] of the tile; rows of length ldm halves.
  const int m = lane & 15;
  const int koff = (lane >> 4) << 4;
  const _Float16* p = base + m * ldm + koff;
  v8h lo = *(const v8h*)(p);        // K = koff .. koff+7
  v8h hi = *(const v8h*)(p + 8);    // K = 8+koff .. 15+koff
  return __builtin_shufflevector(lo, hi, 0, 1, 2, 3, 4, 5, 6, 7,
                                 8, 9, 10, 11, 12, 13, 14, 15);
}

__device__ __forceinline__ v16h load_b_f32(const float* __restrict__ B,
                                           int ldk, int lane) {
  const int nn = lane & 15;
  const int koff = (lane >> 4) << 4;
  v16h b;
  #pragma unroll
  for (int e = 0; e < 16; ++e)
    b[e] = (_Float16)B[(size_t)(koff + e) * ldk + nn];
  return b;
}

__device__ __forceinline__ v16h load_b_lds16(const _Float16* __restrict__ B,
                                             int ldk, int lane) {
  const int nn = lane & 15;
  const int koff = (lane >> 4) << 4;
  v16h b;
  #pragma unroll
  for (int e = 0; e < 16; ++e)
    b[e] = B[(koff + e) * ldk + nn];
  return b;
}

// ---------------------------------------------------------------------------
// Kernel 3: both branches via WMMA. One block per batch n (8 waves).
// Weights are staged zero-padded into LDS with uniform control flow so the
// fragment loads are branch-free vector LDS reads (no EXEC ladders).
// GEMM1: y(16pad x 112) = w1(16 x 256) x pool(256 x 112); BN+ReLU fused into
//        the D-tile store into the LDS B-operand (32pad x 128pad, f16).
// GEMM2: ah = sigmoid(wh(256 x 32pad) x yh + bh), aw likewise -> scratch.
// ---------------------------------------------------------------------------
__global__ __launch_bounds__(256) void branch_kernel(
    const float* __restrict__ pmean, const float* __restrict__ pmax,
    const float* __restrict__ w1, const float* __restrict__ b1,
    const float* __restrict__ gamma, const float* __restrict__ beta,
    const float* __restrict__ rmean, const float* __restrict__ rvar,
    const float* __restrict__ wh, const float* __restrict__ bh,
    const float* __restrict__ ww, const float* __restrict__ bw,
    float* __restrict__ ah, float* __restrict__ aw) {
  __shared__ __align__(16) _Float16 w1p[16][256];    //  8 KB, rows 8..15 = 0
  __shared__ __align__(16) _Float16 whp[256][32];    // 16 KB, cols 8..31 = 0
  __shared__ __align__(16) _Float16 wwp[256][32];    // 16 KB, cols 8..31 = 0
  __shared__ __align__(16) _Float16 yB[2][32][128];  // 16 KB [br][Kpad][Lpad]
  const int n    = blockIdx.x;
  const int tid  = threadIdx.x;
  const int wid  = tid >> 5;
  const int lane = tid & 31;

  // ---- zero everything (uniform) ----
  _Float16* z0 = (_Float16*)w1p;
  for (int i = tid; i < 16 * 256; i += 256) z0[i] = (_Float16)0.f;
  _Float16* z1 = (_Float16*)whp;
  for (int i = tid; i < 256 * 32; i += 256) z1[i] = (_Float16)0.f;
  _Float16* z2 = (_Float16*)wwp;
  for (int i = tid; i < 256 * 32; i += 256) z2[i] = (_Float16)0.f;
  _Float16* z3 = (_Float16*)yB;
  for (int i = tid; i < 2 * 32 * 128; i += 256) z3[i] = (_Float16)0.f;
  __syncthreads();
  // ---- copy valid regions unconditionally (uniform trip counts) ----
  for (int i = tid; i < MIPC * CC; i += 256)          // w1 (8,256)
    w1p[i >> 8][i & 255] = (_Float16)w1[i];
  for (int i = tid; i < CC * MIPC; i += 256) {        // wh/ww (256,8)
    whp[i >> 3][i & 7] = (_Float16)wh[i];
    wwp[i >> 3][i & 7] = (_Float16)ww[i];
  }
  __syncthreads();

  // ---- GEMM1: 2 branches x 7 N-tiles, 8 k-steps each ----
  for (int job = wid; job < 14; job += 8) {
    const int b  = job / 7;
    const int nt = job % 7;
    const int l0 = nt * 16;
    const float* pool = (b == 0 ? pmean : pmax) + (size_t)n * CC * LL;
    v8f acc = {};
    #pragma unroll
    for (int kt = 0; kt < 8; ++kt) {
      v16h A = load_a_lds(&w1p[0][kt * 32], 256, lane);
      v16h B = load_b_f32(pool + (size_t)(kt * 32) * LL + l0, LL, lane);
      acc = __builtin_amdgcn_wmma_f32_16x16x32_f16(
          false, A, false, B, (short)0, acc, false, false);
    }
    // BN + ReLU, store into GEMM2 B-operand (orientation [K=m][N=l] matches)
    const int col = l0 + (lane & 15);
    const int mb  = (lane >> 4) * 8;
    #pragma unroll
    for (int r = 0; r < 8; ++r) {
      int m = mb + r;
      if (m < MIPC) {
        float v = acc[r] + b1[m];
        v = (v - rmean[m]) * (gamma[m] * rsqrtf(rvar[m] + EPSB)) + beta[m];
        v = fmaxf(v, 0.f);
        yB[b][m][col] = (_Float16)v;
      }
    }
  }
  __syncthreads();

  // ---- GEMM2: 2 branches x {h,w} x 16 M-tiles x 4 N-tiles = 256 jobs ----
  for (int job = wid; job < 256; job += 8) {
    const int b   = job >> 7;
    const int rem = job & 127;
    const int hw  = rem >> 6;
    const int mt  = (rem >> 2) & 15;
    const int nt  = rem & 3;
    const int l0  = hw * 56 + nt * 16;    // yh: cols 0..55, yw: cols 56..111
    const _Float16* Wp = (hw == 0 ? &whp[0][0] : &wwp[0][0]);
    const float* bias  = (hw == 0 ? bh : bw);
    float*       dst   = (hw == 0 ? ah : aw);
    v16h A = load_a_lds(Wp + mt * 16 * 32, 32, lane);
    v16h B = load_b_lds16(&yB[b][0][l0], 128, lane);
    v8f acc = {};
    acc = __builtin_amdgcn_wmma_f32_16x16x32_f16(
        false, A, false, B, (short)0, acc, false, false);
    const int col = l0 + (lane & 15);
    const int pos = col - hw * 56;        // position within the 56 range
    const int c0  = mt * 16 + ((lane >> 4) * 8);
    if (pos < 56) {
      #pragma unroll
      for (int r = 0; r < 8; ++r) {
        int c = c0 + r;
        float v = acc[r] + bias[c];
        v = 1.f / (1.f + __expf(-v));
        dst[(((size_t)b * NB + n) * CC + c) * HH + pos] = v;
      }
    }
  }
}

// ---------------------------------------------------------------------------
// Kernel 4: csum[n,c] = mean(x * attn). attn recomputed from rank-1 factors
// so the 103 MB 'out' intermediate never hits memory. x should be L2-hot.
// ---------------------------------------------------------------------------
__global__ __launch_bounds__(256) void mulsum_kernel(
    const float* __restrict__ x,
    const float* __restrict__ ah, const float* __restrict__ aw,
    const float* __restrict__ g, float* __restrict__ csum) {
  __shared__ float fh[4][HH];   // 0=ah_mean 1=ah_max 2=aw_mean 3=aw_max
  __shared__ float red[256];
  const int nc = blockIdx.x;
  const int n = nc / CC, c = nc % CC;
  const int tid = threadIdx.x;
  if (tid < 4 * HH) {
    int j = tid / HH, t = tid % HH;
    const float* src = (j < 2 ? ah : aw);
    int b = j & 1;
    fh[j][t] = src[(((size_t)b * NB + n) * CC + c) * HH + t];
  }
  __syncthreads();
  const float g0 = g[n * 2 + 0], g1 = g[n * 2 + 1];
  const float4* xb4 = (const float4*)(x + (size_t)nc * HW);
  float part = 0.f;
  for (int j = tid; j < HW / 4; j += 256) {
    __builtin_prefetch(xb4 + j + 256, 0, 0);
    float4 v = xb4[j];
    int h = (4 * j) / WWID, w0 = (4 * j) % WWID;
    float am = g0 * fh[0][h], ax = g1 * fh[1][h];
    part += v.x * (am * fh[2][w0 + 0] + ax * fh[3][w0 + 0]);
    part += v.y * (am * fh[2][w0 + 1] + ax * fh[3][w0 + 1]);
    part += v.z * (am * fh[2][w0 + 2] + ax * fh[3][w0 + 2]);
    part += v.w * (am * fh[2][w0 + 3] + ax * fh[3][w0 + 3]);
  }
  red[tid] = part;
  __syncthreads();
  for (int s = 128; s > 0; s >>= 1) {
    if (tid < s) red[tid] += red[tid + s];
    __syncthreads();
  }
  if (tid == 0) csum[nc] = red[0] * (1.f / HW);
}

// ---------------------------------------------------------------------------
// Kernel 5: s = sigmoid(relu(csum @ cw1.T) @ cw2.T + cb2)
// ---------------------------------------------------------------------------
__global__ __launch_bounds__(256) void se_kernel(
    const float* __restrict__ csum,
    const float* __restrict__ cw1, const float* __restrict__ cw2,
    const float* __restrict__ cb2, float* __restrict__ s) {
  __shared__ float hid[64];
  const int n = blockIdx.x, tid = threadIdx.x;
  if (tid < 64) {
    float a = 0.f;
    for (int c = 0; c < CC; ++c) a += csum[n * CC + c] * cw1[tid * CC + c];
    hid[tid] = fmaxf(a, 0.f);
  }
  __syncthreads();
  float a = cb2[tid];
  for (int o = 0; o < 64; ++o) a += hid[o] * cw2[tid * 64 + o];
  s[n * CC + tid] = 1.f / (1.f + __expf(-a));
}

// ---------------------------------------------------------------------------
// Kernel 6: out = x * (attn * s + 1). Reads x (L2), writes out (HBM).
// ---------------------------------------------------------------------------
__global__ __launch_bounds__(256) void final_kernel(
    const float* __restrict__ x,
    const float* __restrict__ ah, const float* __restrict__ aw,
    const float* __restrict__ g, const float* __restrict__ s,
    float* __restrict__ out) {
  __shared__ float fh[4][HH];
  const int nc = blockIdx.x;
  const int n = nc / CC, c = nc % CC;
  const int tid = threadIdx.x;
  if (tid < 4 * HH) {
    int j = tid / HH, t = tid % HH;
    const float* src = (j < 2 ? ah : aw);
    int b = j & 1;
    fh[j][t] = src[(((size_t)b * NB + n) * CC + c) * HH + t];
  }
  __syncthreads();
  const float g0 = g[n * 2 + 0], g1 = g[n * 2 + 1], sv = s[nc];
  const float4* xb4 = (const float4*)(x + (size_t)nc * HW);
  float4* ob4 = (float4*)out + (size_t)nc * (HW / 4);
  for (int j = tid; j < HW / 4; j += 256) {
    __builtin_prefetch(xb4 + j + 256, 0, 0);
    float4 v = xb4[j];
    int h = (4 * j) / WWID, w0 = (4 * j) % WWID;
    float am = g0 * fh[0][h], ax = g1 * fh[1][h];
    float4 o;
    o.x = v.x * ((am * fh[2][w0 + 0] + ax * fh[3][w0 + 0]) * sv + 1.f);
    o.y = v.y * ((am * fh[2][w0 + 1] + ax * fh[3][w0 + 1]) * sv + 1.f);
    o.z = v.z * ((am * fh[2][w0 + 2] + ax * fh[3][w0 + 2]) * sv + 1.f);
    o.w = v.w * ((am * fh[2][w0 + 3] + ax * fh[3][w0 + 3]) * sv + 1.f);
    ob4[j] = o;
  }
}

// ---------------------------------------------------------------------------
extern "C" void kernel_launch(void* const* d_in, const int* in_sizes, int n_in,
                              void* d_out, int out_size, void* d_ws, size_t ws_size,
                              hipStream_t stream) {
  const float* x     = (const float*)d_in[0];
  const float* w1    = (const float*)d_in[1];
  const float* b1    = (const float*)d_in[2];
  const float* gamma = (const float*)d_in[3];
  const float* beta  = (const float*)d_in[4];
  const float* rmean = (const float*)d_in[5];
  const float* rvar  = (const float*)d_in[6];
  const float* wh    = (const float*)d_in[7];
  const float* bh    = (const float*)d_in[8];
  const float* ww    = (const float*)d_in[9];
  const float* bw    = (const float*)d_in[10];
  const float* gw1   = (const float*)d_in[11];
  const float* gw2   = (const float*)d_in[12];
  const float* gb2   = (const float*)d_in[13];
  const float* cw1   = (const float*)d_in[14];
  const float* cw2   = (const float*)d_in[15];
  const float* cb2   = (const float*)d_in[16];
  float* out = (float*)d_out;

  float* ws = (float*)d_ws;
  size_t o = 0;
  float* pmean = ws + o; o += (size_t)NB * CC * LL;        // 917504
  float* pmax  = ws + o; o += (size_t)NB * CC * LL;        // 917504
  float* gap   = ws + o; o += (size_t)NB * CC;             // 8192
  float* ah    = ws + o; o += (size_t)2 * NB * CC * HH;    // 917504
  float* aw    = ws + o; o += (size_t)2 * NB * CC * HH;    // 917504
  float* g     = ws + o; o += (size_t)NB * 2;              // 64
  float* csum  = ws + o; o += (size_t)NB * CC;             // 8192
  float* sbuf  = ws + o; o += (size_t)NB * CC;             // 8192
  (void)ws_size; (void)in_sizes; (void)n_in; (void)out_size;

  pool_kernel  <<<NB * CC, 256, 0, stream>>>(x, pmean, pmax, gap);
  gate_kernel  <<<NB,       64, 0, stream>>>(gap, gw1, gw2, gb2, g);
  branch_kernel<<<NB,      256, 0, stream>>>(pmean, pmax, w1, b1, gamma, beta,
                                             rmean, rvar, wh, bh, ww, bw, ah, aw);
  mulsum_kernel<<<NB * CC, 256, 0, stream>>>(x, ah, aw, g, csum);
  se_kernel    <<<NB,      256, 0, stream>>>(csum, cw1, cw2, cb2, sbuf);
  final_kernel <<<NB * CC, 256, 0, stream>>>(x, ah, aw, g, sbuf, out);
}
